// ArcFace_78512002171447
// MI455X (gfx1250) — matrix operand order repaired and checked by reading, not compiled, not run on hardware
//
#include <hip/hip_runtime.h>
#include <hip/hip_bf16.h>

// ---------------------------------------------------------------------------
// ArcFace forward for MI455X (gfx1250, wave32, WMMA).
//   B=512, E=512, I=100000, S=3, N = I*S = 300000
//   GEMM: C[b,n] = sum_e embf16[b,e] * (w[e,n] * invnorm[e, n%3])   (f16 in, f32 acc)
//   epilogue: max over s, acos, +margin on target, cos*64 -> logits -> log_softmax
// ---------------------------------------------------------------------------

typedef __attribute__((ext_vector_type(16))) _Float16 v16h;
typedef __attribute__((ext_vector_type(8)))  float    v8f;

#define BATCH 512
#define EDIM  512
#define IDENT 100000
#define SSUB  3
#define NCOL  (IDENT * SSUB)      // 300000
#define MARGIN_F 0.5f
#define SCALE_F  64.0f

// GEMM tiling
#define TM 128
#define TN 96                     // 32 identities * 3 subclasses
#define TK 32
#define NKSTEP (EDIM / TK)        // 16
#define ASTR 40                   // LDS halves stride (80B, 16B aligned, bank-spread)
#define BSTR 40
#define CSTR 100                  // LDS floats stride for C staging

// LDS layout (bytes), double buffered A/B, all aliased with C staging:
//   Abuf[p] : p*10240            (TM*ASTR*2   = 10240 each)
//   Bbuf[p] : 20480 + p*7680     (TN*BSTR*2   =  7680 each)
//   invn    : 35840              (1536 floats =  6144)
//   Cs      : 0                  (TM*CSTR*4   = 51200, used only after K loop)
#define SMEM_BYTES (TM * CSTR * 4)

// ---------------------------------------------------------------------------
// Kernel 1: invnorm[e*3+s] = 1 / max(||w[e,:,s]||_2, 1e-12)
// One block per e; float4 reads; (4j+c)%3 == (j+c)%3 -> rotate accumulators.
// ---------------------------------------------------------------------------
__global__ __launch_bounds__(256) void k_invnorm(const float* __restrict__ w,
                                                 float* __restrict__ invn) {
    const int e = blockIdx.x;
    const int t = threadIdx.x;
    const float4* row4 = (const float4*)(w + (size_t)e * NCOL);   // 75000 vec4
    float a0 = 0.f, a1 = 0.f, a2 = 0.f;
    for (int j = t; j < NCOL / 4; j += 256) {
        float4 v = row4[j];
        float p0 = v.x * v.x + v.w * v.w;   // comps 0,3 -> s = j%3
        float p1 = v.y * v.y;               // s = (j+1)%3
        float p2 = v.z * v.z;               // s = (j+2)%3
        int ph = j % 3;
        if (ph == 0)      { a0 += p0; a1 += p1; a2 += p2; }
        else if (ph == 1) { a1 += p0; a2 += p1; a0 += p2; }
        else              { a2 += p0; a0 += p1; a1 += p2; }
    }
    __shared__ float red[3][256];
    red[0][t] = a0; red[1][t] = a1; red[2][t] = a2;
    __syncthreads();
    for (int off = 128; off > 0; off >>= 1) {
        if (t < off) {
            red[0][t] += red[0][t + off];
            red[1][t] += red[1][t + off];
            red[2][t] += red[2][t + off];
        }
        __syncthreads();
    }
    if (t < 3) {
        float n = sqrtf(red[t][0]);
        invn[e * 3 + t] = 1.0f / fmaxf(n, 1e-12f);
    }
}

// ---------------------------------------------------------------------------
// Kernel 2: embedding f32 -> f16 (A matrix for WMMA)
// ---------------------------------------------------------------------------
__global__ __launch_bounds__(256) void k_cvtA(const float* __restrict__ emb,
                                              _Float16* __restrict__ a16) {
    int idx = blockIdx.x * 256 + threadIdx.x;
    if (idx < BATCH * EDIM) a16[idx] = (_Float16)emb[idx];
}

// ---------------------------------------------------------------------------
// Kernel 3: double-buffered WMMA GEMM + fused ArcFace epilogue.
// Grid: (NCOL/TN = 3125, BATCH/TM = 4), 256 threads (8 waves).
// Each wave owns a 32(M) x 48(N) sub-tile = 2x3 fragments of 16x16 f32.
// A tile streamed via global_load_async_to_lds_b128 (ASYNCcnt);
// B tile via float4 register staging + scale/cvt + transposed ds_store.
// ---------------------------------------------------------------------------
__global__ __launch_bounds__(256) void k_gemm_arcface(
    const _Float16* __restrict__ A,      // (BATCH, EDIM) f16 row-major
    const float*    __restrict__ w,      // (EDIM, NCOL) f32 row-major
    const float*    __restrict__ invn,   // (EDIM, 3)
    const float*    __restrict__ target, // (BATCH, IDENT) one-hot f32
    float*          __restrict__ out)    // (BATCH, IDENT) logits
{
    __shared__ __align__(16) unsigned char smem[SMEM_BYTES];
    float* invnLds = (float*)(smem + 35840);
    float* Cs      = (float*)smem;

    const int t    = threadIdx.x;
    const int lane = t & 31;
    const int wave = t >> 5;        // 0..7
    const int wm   = wave & 3;      // M group: 32 rows
    const int wn   = wave >> 2;     // N group: 48 cols
    const int n0   = blockIdx.x * TN;
    const int m0   = blockIdx.y * TM;

    // per-thread fixed tile coordinates
    const int rA   = t >> 1;              // A tile row 0..127
    const int cA   = (t & 1) << 4;        // A tile half-col 0/16
    const int rB   = t >> 3;              // B tile K-row 0..31
    const int cB   = (t & 7) * 12;        // B tile col base (multiple of 12 -> %3 == 0)

    const _Float16* aSrc = A + (size_t)(m0 + rA) * EDIM + cA;  // + k0
    const float*    wSrc = w + (size_t)rB * NCOL + n0 + cB;    // + k0*NCOL

    // ---- helpers -----------------------------------------------------------
    auto issueA = [&](int k0, int p) {
        _Float16* dst = (_Float16*)(smem + p * 10240) + rA * ASTR + cA;
        const _Float16* src = aSrc + k0;
        // CDNA5 async global->LDS copy, 16B per lane, tracked by ASYNCcnt.
        asm volatile("global_load_async_to_lds_b128 %0, %1, off"
                     :: "v"((unsigned)(unsigned long long)(void*)dst),
                        "v"((const void*)src)
                     : "memory");
        asm volatile("global_load_async_to_lds_b128 %0, %1, off"
                     :: "v"((unsigned)(unsigned long long)(void*)(dst + 8)),
                        "v"((const void*)(src + 8))
                     : "memory");
    };
    auto loadB = [&](int k0, float4& b0, float4& b1, float4& b2) {
        const float4* gp = (const float4*)(wSrc + (size_t)k0 * NCOL);
        b0 = gp[0]; b1 = gp[1]; b2 = gp[2];
    };
    auto storeB = [&](int k0, int p, float4 b0, float4 b1, float4 b2) {
        float s0 = invnLds[(k0 + rB) * 3 + 0];
        float s1 = invnLds[(k0 + rB) * 3 + 1];
        float s2 = invnLds[(k0 + rB) * 3 + 2];
        _Float16* bb = (_Float16*)(smem + 20480 + p * 7680) + cB * BSTR + rB;
        bb[0 * BSTR]  = (_Float16)(b0.x * s0);
        bb[1 * BSTR]  = (_Float16)(b0.y * s1);
        bb[2 * BSTR]  = (_Float16)(b0.z * s2);
        bb[3 * BSTR]  = (_Float16)(b0.w * s0);
        bb[4 * BSTR]  = (_Float16)(b1.x * s1);
        bb[5 * BSTR]  = (_Float16)(b1.y * s2);
        bb[6 * BSTR]  = (_Float16)(b1.z * s0);
        bb[7 * BSTR]  = (_Float16)(b1.w * s1);
        bb[8 * BSTR]  = (_Float16)(b2.x * s2);
        bb[9 * BSTR]  = (_Float16)(b2.y * s0);
        bb[10 * BSTR] = (_Float16)(b2.z * s1);
        bb[11 * BSTR] = (_Float16)(b2.w * s2);
    };

    const v8f vzero = {0.f, 0.f, 0.f, 0.f, 0.f, 0.f, 0.f, 0.f};
    v8f acc[2][3];
#pragma unroll
    for (int fm = 0; fm < 2; ++fm)
#pragma unroll
        for (int fn = 0; fn < 3; ++fn) acc[fm][fn] = vzero;

    union F16x16 { v16h h; uint4 q[2]; };
    const int mrow  = lane & 15;
    const int krowA = (lane >> 4) << 3;    // A: K halves {0..7}/{8..15} per lane group
    const int krowB = (lane >> 4) << 3;    // B: K halves {0..7}/{8..15} per lane group

    auto compute = [&](int p) {
        const _Float16* Ab = (const _Float16*)(smem + p * 10240);
        const _Float16* Bb = (const _Float16*)(smem + 20480 + p * 7680);
        F16x16 af[2], bf[3];
#pragma unroll
        for (int fm = 0; fm < 2; ++fm) {
            int row = wm * 32 + fm * 16 + mrow;
            af[fm].q[0] = *(const uint4*)&Ab[row * ASTR + krowA];        // K 0-7 / 8-15
            af[fm].q[1] = *(const uint4*)&Ab[row * ASTR + krowA + 16];   // K 16-23 / 24-31
        }
#pragma unroll
        for (int fn = 0; fn < 3; ++fn) {
            int col = wn * 48 + fn * 16 + mrow;
            bf[fn].q[0] = *(const uint4*)&Bb[col * BSTR + krowB];        // K 0-7 / 8-15
            bf[fn].q[1] = *(const uint4*)&Bb[col * BSTR + krowB + 16];   // K 16-23 / 24-31
        }
#pragma unroll
        for (int fm = 0; fm < 2; ++fm)
#pragma unroll
            for (int fn = 0; fn < 3; ++fn)
                acc[fm][fn] = __builtin_amdgcn_wmma_f32_16x16x32_f16(
                    false, af[fm].h, false, bf[fn].h,
                    (short)0, acc[fm][fn], false, false);
    };

    // ---- prologue: invnorm scales + first tile pair ------------------------
    for (int i = t; i < EDIM * 3; i += 256) invnLds[i] = invn[i];
    issueA(0, 0);
    float4 pb0, pb1, pb2;
    loadB(0, pb0, pb1, pb2);
    __syncthreads();                       // invnLds visible
    storeB(0, 0, pb0, pb1, pb2);
    asm volatile("s_wait_asynccnt 0" ::: "memory");
    __syncthreads();                       // buffer 0 ready

    // ---- pipelined K loop --------------------------------------------------
    for (int kk = 0; kk < NKSTEP; ++kk) {
        const int k0   = kk * TK;
        const int p    = kk & 1;
        const bool more = (kk + 1 < NKSTEP);
        float4 nb0, nb1, nb2;
        if (more) {
            issueA(k0 + TK, p ^ 1);        // async A -> other buffer
            loadB(k0 + TK, nb0, nb1, nb2); // B into registers (latency hidden)
            if (k0 + 2 * TK < EDIM)
                __builtin_prefetch(wSrc + (size_t)(k0 + 2 * TK) * NCOL, 0, 1);
        }
        compute(p);                        // ds_load fragments + 6x WMMA
        if (more) {
            storeB(k0 + TK, p ^ 1, nb0, nb1, nb2);
            asm volatile("s_wait_asynccnt 0" ::: "memory");
        }
        __syncthreads();
    }

    // ---- stage C tile to LDS (VGPR r -> M = r + 8*(lane/16), N = lane%16) --
    union CF { v8f v; float f[8]; };
#pragma unroll
    for (int fm = 0; fm < 2; ++fm)
#pragma unroll
        for (int fn = 0; fn < 3; ++fn) {
            CF cf; cf.v = acc[fm][fn];
            int rbase = wm * 32 + fm * 16 + ((lane >> 4) << 3);
            int col   = wn * 48 + fn * 16 + (lane & 15);
#pragma unroll
            for (int r = 0; r < 8; ++r)
                Cs[(rbase + r) * CSTR + col] = cf.f[r];
        }
    __syncthreads();

    // ---- fused ArcFace epilogue: max over S, acos, margin, cos*scale ----
    for (int idx = t; idx < TM * (TN / 3); idx += 256) {
        int row = idx >> 5;                 // 0..127
        int ic  = idx & 31;                 // identity within tile
        float c0 = Cs[row * CSTR + ic * 3 + 0];
        float c1 = Cs[row * CSTR + ic * 3 + 1];
        float c2 = Cs[row * CSTR + ic * 3 + 2];
        float cosv = fmaxf(c0, fmaxf(c1, c2));
        cosv = fminf(1.0f, fmaxf(-1.0f, cosv));
        int b  = m0 + row;
        int gi = n0 / 3 + ic;
        float tgt   = target[(size_t)b * IDENT + gi];
        float theta = acosf(cosv);
        float logit = cosf(theta + MARGIN_F * tgt) * SCALE_F;
        out[(size_t)b * IDENT + gi] = logit;
    }
}

// ---------------------------------------------------------------------------
// Kernel 4: in-place log_softmax along I (one block per batch row, 3 passes,
// float4 vectorized: 100000 = 25000 * 4)
// ---------------------------------------------------------------------------
__global__ __launch_bounds__(512) void k_logsoftmax(float* __restrict__ out) {
    const int b = blockIdx.x;
    const int t = threadIdx.x;
    float4* row4 = (float4*)(out + (size_t)b * IDENT);   // 25000 vec4
    __shared__ float red[512];

    float m = -INFINITY;
    for (int i = t; i < IDENT / 4; i += 512) {
        float4 v = row4[i];
        m = fmaxf(m, fmaxf(fmaxf(v.x, v.y), fmaxf(v.z, v.w)));
    }
    red[t] = m; __syncthreads();
    for (int off = 256; off > 0; off >>= 1) {
        if (t < off) red[t] = fmaxf(red[t], red[t + off]);
        __syncthreads();
    }
    m = red[0]; __syncthreads();

    float s = 0.f;
    for (int i = t; i < IDENT / 4; i += 512) {
        float4 v = row4[i];
        s += expf(v.x - m) + expf(v.y - m) + expf(v.z - m) + expf(v.w - m);
    }
    red[t] = s; __syncthreads();
    for (int off = 256; off > 0; off >>= 1) {
        if (t < off) red[t] += red[t + off];
        __syncthreads();
    }
    float rc = m + logf(red[0]);

    for (int i = t; i < IDENT / 4; i += 512) {
        float4 v = row4[i];
        v.x -= rc; v.y -= rc; v.z -= rc; v.w -= rc;
        row4[i] = v;
    }
}

// ---------------------------------------------------------------------------
extern "C" void kernel_launch(void* const* d_in, const int* in_sizes, int n_in,
                              void* d_out, int out_size, void* d_ws, size_t ws_size,
                              hipStream_t stream) {
    const float* emb    = (const float*)d_in[0];   // (512, 512)
    const float* target = (const float*)d_in[1];   // (512, 100000)
    const float* w      = (const float*)d_in[2];   // (512, 100000, 3)
    float* out = (float*)d_out;                    // (512, 100000)

    // workspace layout: invnorm (6KB, rounded to 8KB) | A f16 (512KB)
    float*    invn = (float*)d_ws;
    _Float16* a16  = (_Float16*)((char*)d_ws + 8192);

    k_invnorm<<<EDIM, 256, 0, stream>>>(w, invn);
    k_cvtA<<<(BATCH * EDIM) / 256, 256, 0, stream>>>(emb, a16);

    dim3 grid(NCOL / TN, BATCH / TM);   // (3125, 4)
    k_gemm_arcface<<<grid, 256, 0, stream>>>(a16, w, invn, target, out);

    k_logsoftmax<<<BATCH, 512, 0, stream>>>(out);
}
